// SemanticDriftCoeff_2448131359021
// MI455X (gfx1250) — matrix operand order repaired and checked by reading, compiled, not run on hardware
//
#include <hip/hip_runtime.h>
#include <hip/hip_bf16.h>

typedef __bf16 bf16_t;
typedef __attribute__((ext_vector_type(16))) __bf16 v16bf;
typedef __attribute__((ext_vector_type(8)))  float  v8f;
typedef __attribute__((ext_vector_type(4)))  unsigned int v4u;
typedef __attribute__((ext_vector_type(8)))  int v8i;
typedef __attribute__((ext_vector_type(4)))  int v4i;

#define T_LEN 4096
#define D_DIM 896
#define D_AUG 912          // 896 data cols + col 896 == 1.0 (row-sum trick) + zero pad
#define KCH   28           // 896 / 32 k-chunks for sim / cross
#define SCH   256          // s-chunk width in k_attention
#define OMEGA_C 0.3f
#define NU_C    0.2f
#define LAM_C   0.1f
#define SQRT_D  29.9332590942f   // sqrt(896)

__device__ inline v8f zero8() {
  v8f z = {0.f, 0.f, 0.f, 0.f, 0.f, 0.f, 0.f, 0.f};
  return z;
}

// ---- Tensor Data Mover: DMA a rows x width bf16 tile (row stride = stride elems)
// from global into LDS. D# layout per CDNA5 ISA ch.8 (group0/group1, 2D tile).
// This toolchain exposes the 6-arg builtin: (g0, g1, g2, g3, g4, cpol).
__device__ inline void tdm_load_tile(void* lds_dst, const bf16_t* gsrc,
                                     int rows, int width, int stride) {
  unsigned long long ga = (unsigned long long)(uintptr_t)gsrc;
  unsigned lds_off = (unsigned)(unsigned long long)(uintptr_t)lds_dst; // flat[31:0] == LDS offset
  const unsigned DIM1 = 1u << 20;          // tensor_dim1: large (tile fully in-bounds)
  v4u g0;
  g0[0] = 1u;                                               // count=1, user descriptor
  g0[1] = lds_off;                                          // lds_addr
  g0[2] = (unsigned)(ga & 0xffffffffu);                     // global_addr[31:0]
  g0[3] = (unsigned)((ga >> 32) & 0x01ffffffu) | (2u << 30);// global_addr[56:32] | type=2
  v8i g1;
  g1[0] = (int)(1u << 16);                                  // data_size=1 (2B); no pad/iter/mcast
  g1[1] = (int)(((unsigned)width & 0xffffu) << 16);         // tensor_dim0[15:0] @ bits63:48
  g1[2] = (int)((((unsigned)width >> 16) & 0xffffu)         // tensor_dim0[31:16]
              | ((DIM1 & 0xffffu) << 16));                  // tensor_dim1[15:0]
  g1[3] = (int)(((DIM1 >> 16) & 0xffffu)                    // tensor_dim1[31:16]
              | (((unsigned)width & 0xffffu) << 16));       // tile_dim0
  g1[4] = rows & 0xffff;                                    // tile_dim1 (tile_dim2=0)
  g1[5] = stride;                                           // tensor_dim0_stride[31:0]
  g1[6] = 0;                                                // stride hi | dim1_stride lo
  g1[7] = 0;
  v4i z4 = {0, 0, 0, 0};
  v8i z8 = {0, 0, 0, 0, 0, 0, 0, 0};
  __builtin_amdgcn_tensor_load_to_lds(g0, g1, z4, z4, z8, 0);
}

// A operand (16x32 bf16): lanes 0-15 -> M=lane, lanes 16-31 -> M=lane-16.
__device__ inline v16bf load_a_tile(const bf16_t* base, int stride, int m, int half, int k0) {
  v16bf a;
  const bf16_t* p = base + (size_t)m * stride + k0 + half * 16;
#pragma unroll
  for (int i = 0; i < 16; ++i) a[i] = p[i];
  return a;
}

// B operand (32x16 bf16) where B[k][n] = src[s0+n][k0+k]  -> contiguous 32B load.
__device__ inline v16bf load_b_rows(const bf16_t* __restrict__ hb, int s0, int n, int khalf, int k0) {
  v16bf b;
  const bf16_t* p = hb + (size_t)(s0 + n) * D_AUG + k0 + khalf * 16;
#pragma unroll
  for (int i = 0; i < 16; ++i) b[i] = p[i];
  return b;
}

// B operand from TRANSPOSED copy: B[k][n] = hbT[d0+n][srow0+k]  -> contiguous 32B load.
__device__ inline v16bf load_b_T(const bf16_t* __restrict__ hbT, int d, int khalf, int srow0) {
  v16bf b;
  const bf16_t* p = hbT + (size_t)d * T_LEN + srow0 + khalf * 16;
#pragma unroll
  for (int i = 0; i < 16; ++i) b[i] = p[i];
  return b;
}

__device__ inline v8f wmma_bf16(v16bf a, v16bf b, v8f c) {
  return __builtin_amdgcn_wmma_f32_16x16x32_bf16(false, a, false, b, (short)0, c, false, false);
}

// ---------------- Kernel 1: row-normalize, emit f32 h, bf16 h (augmented), hn2 ----
__global__ void __launch_bounds__(256)
k_normalize(const float* __restrict__ x, float* __restrict__ h,
            bf16_t* __restrict__ hb, float* __restrict__ hn2) {
  __shared__ float red[256];
  int t = blockIdx.x;
  int tid = threadIdx.x;
  const float* xr = x + (size_t)t * D_DIM;
  float ss = 0.f;
  for (int d = tid; d < D_DIM; d += 256) { float v = xr[d]; ss += v * v; }
  red[tid] = ss;
  __syncthreads();
  for (int off = 128; off > 0; off >>= 1) {
    if (tid < off) red[tid] += red[tid + off];
    __syncthreads();
  }
  float sumsq = red[0];
  float rinv = 1.0f / fmaxf(sqrtf(sumsq), 1e-12f);
  for (int d = tid; d < D_DIM; d += 256) {
    float v = xr[d] * rinv;
    h[(size_t)t * D_DIM + d] = v;
    hb[(size_t)t * D_AUG + d] = (bf16_t)v;
  }
  if (tid < 16) hb[(size_t)t * D_AUG + D_DIM + tid] = (bf16_t)((tid == 0) ? 1.0f : 0.0f);
  if (tid == 0) hn2[t] = sumsq * rinv * rinv;
}

// ---------------- Kernel 1b: tiled transpose hb[t][d] -> hbT[d][t] ----------------
__global__ void __launch_bounds__(256)
k_transpose(const bf16_t* __restrict__ hb, bf16_t* __restrict__ hbT) {
  __shared__ bf16_t tile[32][33];
  int tb = blockIdx.x * 32;
  int db = blockIdx.y * 32;
  int tx = threadIdx.x & 31;
  int ty = threadIdx.x >> 5;
  for (int r = ty; r < 32; r += 8) {
    int d = db + tx;
    tile[r][tx] = (d < D_AUG) ? hb[(size_t)(tb + r) * D_AUG + d] : (bf16_t)0.0f;
  }
  __syncthreads();
  for (int r = ty; r < 32; r += 8) {
    int d = db + r;
    if (d < D_AUG) hbT[(size_t)d * T_LEN + tb + tx] = tile[tx][r];
  }
}

// ---------------- Kernel 2: causal decayed-softmax attention, unnormalized --------
// U[t, 0:896] = sum_s w(t,s) h[s,:],  R[t] = sum_s w(t,s)   (via ones column 896)
__global__ void __launch_bounds__(256)
k_attention(const bf16_t* __restrict__ hb, const bf16_t* __restrict__ hbT,
            float* __restrict__ U, float* __restrict__ R) {
  __shared__ bf16_t hm[16 * D_AUG];   // A-rows t0..t0+15 (~29 KB), filled by TDM
  __shared__ bf16_t P[16 * SCH];      // exp-weight tile, bf16 (8 KB)
  int mb = blockIdx.x;
  int t0 = mb * 16;
  int tmax = t0 + 15;
  int tid = threadIdx.x;
  int wave_s = __builtin_amdgcn_readfirstlane(tid >> 5);  // SGPR wave id: scalar branches
  int lane = tid & 31;
  int n = lane & 15;
  int half = lane >> 4;

  // Async DMA the 16 x D_AUG A-tile into LDS with the Tensor Data Mover.
  if (wave_s == 0) {
    tdm_load_tile(hm, hb + (size_t)t0 * D_AUG, 16, D_AUG, D_AUG);
    __builtin_amdgcn_s_wait_tensorcnt(0);
  }
  __syncthreads();

  int wstart = wave_s * 7;
  int wcnt = (wave_s == 7) ? 8 : 7;    // 57 output tiles (incl. row-sum tile) over 8 waves
  v8f acc[8];
#pragma unroll
  for (int i = 0; i < 8; ++i) acc[i] = zero8();

  int nchunks = tmax / SCH + 1;
  for (int ch = 0; ch < nchunks; ++ch) {
    int sbase = ch * SCH;
    if (sbase + SCH <= tmax) {  // warm next chunk's B rows toward the caches
      __builtin_prefetch(hb + (size_t)(sbase + SCH) * D_AUG + (size_t)tid * 64, 0, 0);
    }
    // ---- phase A: two sim tiles per wave sharing each A fetch (scalar guards) ----
    int s0a = sbase + (2 * wave_s) * 16;
    int s0b = s0a + 16;
    bool da = (s0a <= tmax);
    bool db = (s0b <= tmax);
    v8f c0 = zero8(), c1 = zero8();
    if (da) {
      for (int kc = 0; kc < KCH; ++kc) {
        v16bf a = load_a_tile(hm, D_AUG, n, half, kc * 32);
        v16bf b0 = load_b_rows(hb, s0a, n, half, kc * 32);
        c0 = wmma_bf16(a, b0, c0);
        if (db) {
          v16bf b1 = load_b_rows(hb, s0b, n, half, kc * 32);
          c1 = wmma_bf16(a, b1, c1);
        }
      }
    }
    int col0 = (2 * wave_s) * 16;
#pragma unroll
    for (int v = 0; v < 8; ++v) {
      int m = v + half * 8;
      int t = t0 + m;
      int s = s0a + n;
      float w = 0.f;
      if (da && s <= t) {
        float denom = SQRT_D * __expf(LAM_C * (float)(t - s)) + 1e-8f;
        w = __expf(c0[v] * __builtin_amdgcn_rcpf(denom));  // |logit|<=0.034: no max-sub needed
      }
      P[m * SCH + col0 + n] = (bf16_t)w;
      s = s0b + n;
      w = 0.f;
      if (db && s <= t) {
        float denom = SQRT_D * __expf(LAM_C * (float)(t - s)) + 1e-8f;
        w = __expf(c1[v] * __builtin_amdgcn_rcpf(denom));
      }
      P[m * SCH + col0 + 16 + n] = (bf16_t)w;
    }
    __syncthreads();
    // ---- phase B: acc += P @ h_aug[sbase:sbase+256, :] via transposed copy ----
    for (int kc = 0; kc < SCH / 32; ++kc) {
      v16bf a = load_a_tile(P, SCH, n, half, kc * 32);
      int srow0 = sbase + kc * 32;
#pragma unroll
      for (int u = 0; u < 8; ++u) {
        if (u < wcnt) {
          int d = (wstart + u) * 16 + n;
          v16bf b = load_b_T(hbT, d, half, srow0);
          acc[u] = wmma_bf16(a, b, acc[u]);
        }
      }
    }
    __syncthreads();
  }

#pragma unroll
  for (int u = 0; u < 8; ++u) {
    if (u < wcnt) {
      int d = (wstart + u) * 16 + n;
#pragma unroll
      for (int v = 0; v < 8; ++v) {
        int t = t0 + v + half * 8;
        if (d < D_DIM)       U[(size_t)t * D_DIM + d] = acc[u][v];
        else if (d == D_DIM) R[t] = acc[u][v];   // ones-column == softmax row sum
      }
    }
  }
}

// ---------------- Kernel 3: g = U / R, gn2, dist ----------------------------------
__global__ void __launch_bounds__(256)
k_finalize_g(const float* __restrict__ U, const float* __restrict__ R,
             const float* __restrict__ h, bf16_t* __restrict__ gb,
             float* __restrict__ gn2, float* __restrict__ dist) {
  __shared__ float red1[256];
  __shared__ float red2[256];
  int t = blockIdx.x, tid = threadIdx.x;
  float rinv = 1.0f / R[t];
  float sg = 0.f, sd = 0.f;
  for (int d = tid; d < D_DIM; d += 256) {
    float g = U[(size_t)t * D_DIM + d] * rinv;
    gb[(size_t)t * D_DIM + d] = (bf16_t)g;
    float df = h[(size_t)t * D_DIM + d] - g;
    sg += g * g;
    sd += df * df;
  }
  red1[tid] = sg; red2[tid] = sd;
  __syncthreads();
  for (int off = 128; off > 0; off >>= 1) {
    if (tid < off) { red1[tid] += red1[tid + off]; red2[tid] += red2[tid + off]; }
    __syncthreads();
  }
  if (tid == 0) {
    gn2[t] = red1[0];
    dist[t] = sqrtf(fmaxf(red2[0], 1e-24f));
  }
}

// ---------------- Kernel 4: causal dist_max via cross = g h^T, finalize out -------
__global__ void __launch_bounds__(256)
k_distmax(const bf16_t* __restrict__ gb, const bf16_t* __restrict__ hb,
          const float* __restrict__ hn2, const float* __restrict__ gn2,
          const float* __restrict__ dist,
          const float* __restrict__ m_t, const float* __restrict__ c_t,
          const float* __restrict__ d_t, const float* __restrict__ mu_p,
          float* __restrict__ out) {
  __shared__ bf16_t gm[16 * D_DIM];   // g rows t0..t0+15 (28 KB), filled by TDM
  __shared__ float gn2s[16];
  __shared__ int smax[16];
  int mb = blockIdx.x;
  int t0 = mb * 16;
  int tid = threadIdx.x;
  int wave_s = __builtin_amdgcn_readfirstlane(tid >> 5);
  int lane = tid & 31;
  int n = lane & 15, half = lane >> 4;

  if (wave_s == 0) {
    tdm_load_tile(gm, gb + (size_t)t0 * D_DIM, 16, D_DIM, D_DIM);
    __builtin_amdgcn_s_wait_tensorcnt(0);
  }
  if (tid < 16) { gn2s[tid] = gn2[t0 + tid]; smax[tid] = 0; }
  __syncthreads();

  float rm[8];
#pragma unroll
  for (int v = 0; v < 8; ++v) rm[v] = 0.f;

  for (int st = wave_s; st <= mb; st += 8) {   // causal s-tiles striped over waves (scalar loop)
    int s0 = st * 16;
    v8f c = zero8();
    for (int kc = 0; kc < KCH; ++kc) {
      v16bf a = load_a_tile(gm, D_DIM, n, half, kc * 32);
      v16bf b = load_b_rows(hb, s0, n, half, kc * 32);
      c = wmma_bf16(a, b, c);
    }
    int s = s0 + n;
    float h2 = hn2[s];
#pragma unroll
    for (int v = 0; v < 8; ++v) {
      int t = t0 + v + half * 8;
      if (s <= t) {
        float d2 = h2 - 2.0f * c[v] + gn2s[v + half * 8];
        float dm = sqrtf(fmaxf(d2, 1e-24f));
        rm[v] = fmaxf(rm[v], dm);
      }
    }
  }
#pragma unroll
  for (int v = 0; v < 8; ++v)
    atomicMax(&smax[v + half * 8], __float_as_int(rm[v]));  // dist >= 0: int order == float order
  __syncthreads();

  if (tid < 16) {
    int t = t0 + tid;
    float dm = __int_as_float(smax[tid]);
    if (dm < 1e-6f) dm = 1.0f;
    float stab = 1.0f - OMEGA_C * c_t[t] + NU_C * d_t[t];
    float xi = 1.0f - mu_p[0] * m_t[t];
    float val = dist[t] / (dm + 1e-8f) * stab * xi;
    out[t] = fminf(fmaxf(val, 0.0f), 1.0f);
  }
}

extern "C" void kernel_launch(void* const* d_in, const int* in_sizes, int n_in,
                              void* d_out, int out_size, void* d_ws, size_t ws_size,
                              hipStream_t stream) {
  const float* x    = (const float*)d_in[0];   // hidden_states [T, D]
  const float* m_t  = (const float*)d_in[1];
  const float* c_t  = (const float*)d_in[2];
  const float* d_t  = (const float*)d_in[3];
  const float* mu   = (const float*)d_in[4];   // scalar

  char* ws = (char*)d_ws;
  size_t off = 0;
  bf16_t* hb  = (bf16_t*)(ws + off); off += (size_t)T_LEN * D_AUG * sizeof(bf16_t);
  bf16_t* hbT = (bf16_t*)(ws + off); off += (size_t)D_AUG * T_LEN * sizeof(bf16_t);
  float*  h   = (float*) (ws + off); off += (size_t)T_LEN * D_DIM * sizeof(float);
  float*  hn2 = (float*) (ws + off); off += (size_t)T_LEN * sizeof(float);
  float*  U   = (float*) (ws + off); off += (size_t)T_LEN * D_DIM * sizeof(float);
  float*  R   = (float*) (ws + off); off += (size_t)T_LEN * sizeof(float);
  bf16_t* gb  = (bf16_t*)(ws + off); off += (size_t)T_LEN * D_DIM * sizeof(bf16_t);
  float*  gn2 = (float*) (ws + off); off += (size_t)T_LEN * sizeof(float);
  float*  dst = (float*) (ws + off); off += (size_t)T_LEN * sizeof(float);
  float* out = (float*)d_out;

  k_normalize <<<T_LEN, 256, 0, stream>>>(x, h, hb, hn2);
  dim3 tgrid(T_LEN / 32, (D_AUG + 31) / 32);
  k_transpose <<<tgrid, 256, 0, stream>>>(hb, hbT);
  k_attention <<<T_LEN / 16, 256, 0, stream>>>(hb, hbT, U, R);
  k_finalize_g<<<T_LEN, 256, 0, stream>>>(U, R, h, gb, gn2, dst);
  k_distmax   <<<T_LEN / 16, 256, 0, stream>>>(gb, hb, hn2, gn2, dst,
                                               m_t, c_t, d_t, mu, out);
}
